// GATNN_76416058131192
// MI455X (gfx1250) — matrix operand order repaired and checked by reading, not compiled, not run on hardware
//
#include <hip/hip_runtime.h>
#include <hip/hip_bf16.h>

// ---------------------------------------------------------------------------
// GATv2 x2 layers for MI455X (gfx1250, wave32, WMMA).
//   GEMMs: bf16 v_wmma_f32_16x16x32_bf16, weights staged (transposed+converted)
//          in LDS; one kernel produces xl and xr sharing A fragments.
//   Edge phase: wave-per-edge logits + shfl reduce, segment softmax via
//          ordered-key u32 atomic max + f32 atomic add, scatter via f32 atomics.
// ---------------------------------------------------------------------------

typedef __attribute__((ext_vector_type(16))) __bf16 v16bf;
typedef __attribute__((ext_vector_type(8)))  float  v8f;

__device__ __forceinline__ unsigned int pack_bf2(float lo, float hi) {
  // round-to-nearest-even f32 -> bf16, packed pair (lo in [15:0], hi in [31:16])
  unsigned int a = __float_as_uint(lo);
  unsigned int b = __float_as_uint(hi);
  a = (a + (0x7FFFu + ((a >> 16) & 1u))) >> 16;
  b = (b + (0x7FFFu + ((b >> 16) & 1u))) >> 16;
  return (a & 0xFFFFu) | (b << 16);
}

// monotone unsigned key for float total order (for atomic segment-max)
__device__ __forceinline__ unsigned int fkey(float f) {
  unsigned int b = __float_as_uint(f);
  return (b & 0x80000000u) ? ~b : (b | 0x80000000u);
}
__device__ __forceinline__ float funkey(unsigned int k) {
  unsigned int b = (k & 0x80000000u) ? (k & 0x7FFFFFFFu) : ~k;
  return __uint_as_float(b);
}

// ---------------------------------------------------------------------------
// Dual GEMM: outL = X @ Wl + bl ; outR = X @ Wr + br   (X: nRows x 128, W: 128x128)
// Block = 128 threads (4 waves); each wave owns a 16-row strip, 128 cols.
// A fragments (16x32 bf16 per k-step) are loaded once and reused for both mats.
// ---------------------------------------------------------------------------
__global__ __launch_bounds__(128) void gat_gemm_dual(
    const float* __restrict__ X,
    const float* __restrict__ Wl, const float* __restrict__ Wr,
    const float* __restrict__ bl, const float* __restrict__ br,
    float* __restrict__ outL, float* __restrict__ outR, int nRows) {
  // LDS: W transposed, bf16-pair packed, padded stride 68 dwords (bank spread)
  __shared__ __align__(16) unsigned int sW[128 * 68];

  const int tid   = threadIdx.x;
  const int lane  = tid & 31;
  const int wave  = tid >> 5;
  const int half  = lane >> 4;   // 0: lanes 0-15, 1: lanes 16-31
  const int l16   = lane & 15;
  const int mBase = blockIdx.x * 64 + wave * 16;
  const int rClampMax = nRows - 1;

  // ---- A fragments for all 4 k-steps (K = 128 = 4 x 32) ----
  union { v16bf v; unsigned int u[8]; } afrag[4];
  {
    int r = mBase + l16;
    if (r > rClampMax) r = rClampMax;
    const float* xrow = X + (size_t)r * 128;
#pragma unroll
    for (int kt = 0; kt < 4; ++kt) {
#pragma unroll
      for (int v = 0; v < 8; ++v) {
        // 16-bit A layout: VGPR v holds K pair k,k+1 with
        // k = kt*32 + (v<4 ? 0 : 16) + half*8 + (v&3)*2
        int k = kt * 32 + ((v >> 2) << 4) + (half << 3) + ((v & 3) << 1);
        float2 p = *(const float2*)(xrow + k);
        afrag[kt].u[v] = pack_bf2(p.x, p.y);
      }
    }
  }

  for (int mat = 0; mat < 2; ++mat) {
    const float* W    = mat ? Wr : Wl;
    const float* bias = mat ? br : bl;
    float*       out  = mat ? outR : outL;

    __syncthreads();  // previous mat's LDS reads done
    // Transpose + convert W (f32 row-major [k][n]) -> LDS bf16 pairs [n][kpair]
    for (int idx = tid; idx < 128 * 64; idx += 128) {
      int n = idx >> 6, p = idx & 63, k = p << 1;
      sW[n * 68 + p] = pack_bf2(W[k * 128 + n], W[(k + 1) * 128 + n]);
    }
    __syncthreads();

#pragma unroll
    for (int nt = 0; nt < 8; ++nt) {
      const int col = nt * 16 + l16;
      const float bv = bias[col];
      v8f acc = {};
#pragma unroll
      for (int kt = 0; kt < 4; ++kt) {
        // B 32x16 bf16: lane half owns K = kt*32 + half*8 .. +7 and kt*32 + 16 + half*8 .. +7, n = l16
        union { v16bf v; uint4 q[2]; } bfrag;
        const unsigned int* bsrc = &sW[col * 68 + kt * 16 + half * 4];
        bfrag.q[0] = *(const uint4*)(bsrc);
        bfrag.q[1] = *(const uint4*)(bsrc + 8);
        acc = __builtin_amdgcn_wmma_f32_16x16x32_bf16(
            false, afrag[kt].v, false, bfrag.v, (short)0, acc, false, false);
      }
      // C/D layout: VGPR r -> row mBase + r + half*8, col = nt*16 + l16
#pragma unroll
      for (int rr = 0; rr < 8; ++rr) {
        int row = mBase + rr + (half << 3);
        out[(size_t)row * 128 + col] = acc[rr] + bv;
      }
    }
  }
}

// ---------------------------------------------------------------------------
// Per-call init: accum = 0, emaxKey = 0 (== key(-inf) lower bound), denom = 0
// ---------------------------------------------------------------------------
__global__ void gat_init(float* __restrict__ accum,
                         unsigned int* __restrict__ emaxKey,
                         float* __restrict__ denom, int N) {
  int i = blockIdx.x * blockDim.x + threadIdx.x;
  if (i < N * 128) accum[i] = 0.f;
  if (i < N) { emaxKey[i] = 0u; denom[i] = 0.f; }
}

// ---------------------------------------------------------------------------
// Edge logits: wave per edge. e = sum_k leaky(xl[src,k]+xr[dst,k], .2)*att[k]
// Also atomic segment-max of e over dst (ordered-key u32 max).
// Edges >= E are self-loops (src = dst = e - E).
// ---------------------------------------------------------------------------
__global__ __launch_bounds__(256) void gat_edge_logits(
    const float* __restrict__ xl, const float* __restrict__ xr,
    const float* __restrict__ att,
    const long long* __restrict__ srcIdx, const long long* __restrict__ dstIdx,
    float* __restrict__ elog, unsigned int* __restrict__ emaxKey,
    int E, int ET) {
  const int lane = threadIdx.x & 31;
  const int wid = (blockIdx.x * blockDim.x + threadIdx.x) >> 5;
  const int nw  = (gridDim.x * blockDim.x) >> 5;
  float areg[4] = { att[lane], att[lane + 32], att[lane + 64], att[lane + 96] };

  for (int e = wid; e < ET; e += nw) {
    int s, d;
    if (e < E) { s = (int)srcIdx[e]; d = (int)dstIdx[e]; }
    else       { s = e - E;          d = s; }
    const float* pl = xl + (size_t)s * 128 + lane;
    const float* pr = xr + (size_t)d * 128 + lane;
    float sum = 0.f;
#pragma unroll
    for (int j = 0; j < 4; ++j) {
      float t = pl[32 * j] + pr[32 * j];
      t = (t > 0.f) ? t : 0.2f * t;
      sum += t * areg[j];
    }
#pragma unroll
    for (int off = 16; off > 0; off >>= 1)
      sum += __shfl_xor(sum, off, 32);
    if (lane == 0) {
      elog[e] = sum;
      atomicMax(emaxKey + d, fkey(sum));
    }
  }
}

// ---------------------------------------------------------------------------
// ex = exp(e - emax[dst]); denom[dst] += ex   (ex stored in place of e)
// ---------------------------------------------------------------------------
__global__ void gat_edge_exp(float* __restrict__ elog,
                             const unsigned int* __restrict__ emaxKey,
                             float* __restrict__ denom,
                             const long long* __restrict__ dstIdx,
                             int E, int ET) {
  int e = blockIdx.x * blockDim.x + threadIdx.x;
  if (e >= ET) return;
  int d = (e < E) ? (int)dstIdx[e] : (e - E);
  float m  = funkey(emaxKey[d]);
  float ex = __expf(elog[e] - m);
  elog[e] = ex;
  atomicAdd(denom + d, ex);
}

// ---------------------------------------------------------------------------
// Scatter: accum[dst] += xl[src] * (ex / (denom[dst] + 1e-16)); wave per edge.
// ---------------------------------------------------------------------------
__global__ __launch_bounds__(256) void gat_edge_scatter(
    const float* __restrict__ xl, const float* __restrict__ elog,
    const float* __restrict__ denom,
    const long long* __restrict__ srcIdx, const long long* __restrict__ dstIdx,
    float* __restrict__ accum, int E, int ET) {
  const int lane = threadIdx.x & 31;
  const int wid = (blockIdx.x * blockDim.x + threadIdx.x) >> 5;
  const int nw  = (gridDim.x * blockDim.x) >> 5;
  for (int e = wid; e < ET; e += nw) {
    int s, d;
    if (e < E) { s = (int)srcIdx[e]; d = (int)dstIdx[e]; }
    else       { s = e - E;          d = s; }
    float alpha = elog[e] / (denom[d] + 1e-16f);
    const float* pl = xl + (size_t)s * 128 + lane;
    float* pa = accum + (size_t)d * 128 + lane;
#pragma unroll
    for (int j = 0; j < 4; ++j)
      atomicAdd(pa + 32 * j, pl[32 * j] * alpha);
  }
}

// ---------------------------------------------------------------------------
// out = accum + bo  (+ optional leaky_relu(., 0.01) between layers)
// ---------------------------------------------------------------------------
__global__ void gat_finalize(const float* __restrict__ accum,
                             const float* __restrict__ bo,
                             float* __restrict__ out, int N, int applyRelu) {
  int i = blockIdx.x * blockDim.x + threadIdx.x;
  if (i >= N * 128) return;
  float v = accum[i] + bo[i & 127];
  if (applyRelu) v = (v > 0.f) ? v : 0.01f * v;
  out[i] = v;
}

// ---------------------------------------------------------------------------
extern "C" void kernel_launch(void* const* d_in, const int* in_sizes, int n_in,
                              void* d_out, int out_size, void* d_ws, size_t ws_size,
                              hipStream_t stream) {
  (void)n_in; (void)out_size; (void)ws_size;

  const float*     x    = (const float*)d_in[0];
  const long long* ei   = (const long long*)d_in[1];
  const float*     Wl0  = (const float*)d_in[2];
  const float*     bl0  = (const float*)d_in[3];
  const float*     Wr0  = (const float*)d_in[4];
  const float*     br0  = (const float*)d_in[5];
  const float*     att0 = (const float*)d_in[6];
  const float*     bo0  = (const float*)d_in[7];
  const float*     Wl1  = (const float*)d_in[8];
  const float*     bl1  = (const float*)d_in[9];
  const float*     Wr1  = (const float*)d_in[10];
  const float*     br1  = (const float*)d_in[11];
  const float*     att1 = (const float*)d_in[12];
  const float*     bo1  = (const float*)d_in[13];

  const int N  = in_sizes[0] / 128;
  const int E  = in_sizes[1] / 2;
  const int ET = E + N;                    // + self-loops
  const int NP = (N + 63) & ~63;           // padded rows for guard-free tiles

  const long long* srcI = ei;
  const long long* dstI = ei + E;

  // workspace carve-up (256B aligned)
  char* ws = (char*)d_ws;
  size_t off = 0;
  auto carve = [&](size_t bytes) {
    void* p = ws + off;
    off += (bytes + 255) & ~(size_t)255;
    return p;
  };
  float*        xl      = (float*)carve((size_t)NP * 128 * sizeof(float));
  float*        xr      = (float*)carve((size_t)NP * 128 * sizeof(float));
  float*        h       = (float*)carve((size_t)NP * 128 * sizeof(float));
  float*        accum   = (float*)carve((size_t)N * 128 * sizeof(float));
  float*        elog    = (float*)carve((size_t)ET * sizeof(float));
  unsigned int* emaxKey = (unsigned int*)carve((size_t)N * sizeof(unsigned int));
  float*        denom   = (float*)carve((size_t)N * sizeof(float));

  const int gElems = (N * 128 + 255) / 256;   // element-wise grids
  const int gGemm  = NP / 64;                 // 64 rows per block
  const int gEdgeW = (ET + 7) / 8;            // 8 waves per 256-thread block
  const int gEdgeT = (ET + 255) / 256;        // thread-per-edge grid

  // ---------------- layer 0 ----------------
  gat_init<<<gElems, 256, 0, stream>>>(accum, emaxKey, denom, N);
  gat_gemm_dual<<<gGemm, 128, 0, stream>>>(x, Wl0, Wr0, bl0, br0, xl, xr, N);
  gat_edge_logits<<<gEdgeW, 256, 0, stream>>>(xl, xr, att0, srcI, dstI,
                                              elog, emaxKey, E, ET);
  gat_edge_exp<<<gEdgeT, 256, 0, stream>>>(elog, emaxKey, denom, dstI, E, ET);
  gat_edge_scatter<<<gEdgeW, 256, 0, stream>>>(xl, elog, denom, srcI, dstI,
                                               accum, E, ET);
  gat_finalize<<<gElems, 256, 0, stream>>>(accum, bo0, h, N, /*relu=*/1);

  // ---------------- layer 1 ----------------
  gat_init<<<gElems, 256, 0, stream>>>(accum, emaxKey, denom, N);
  gat_gemm_dual<<<gGemm, 128, 0, stream>>>(h, Wl1, Wr1, bl1, br1, xl, xr, N);
  gat_edge_logits<<<gEdgeW, 256, 0, stream>>>(xl, xr, att1, srcI, dstI,
                                              elog, emaxKey, E, ET);
  gat_edge_exp<<<gEdgeT, 256, 0, stream>>>(elog, emaxKey, denom, dstI, E, ET);
  gat_edge_scatter<<<gEdgeW, 256, 0, stream>>>(xl, elog, denom, srcI, dstI,
                                               accum, E, ET);
  gat_finalize<<<gElems, 256, 0, stream>>>(accum, bo1, (float*)d_out, N,
                                           /*relu=*/0);
}
